// Shift_gcn_9122510536898
// MI455X (gfx1250) — matrix operand order repaired and checked by reading, not compiled, not run on hardware
//
#include <hip/hip_runtime.h>
#include <cmath>

// Problem constants (Shift-GCN block)
#define NB      64          // batch
#define TT      256         // frames
#define VV      25          // joints
#define CC      64          // channels (in == out)
#define NTROWS  (NB*TT)     // 16384 rows for BatchNorm
#define ROWLEN  (VV*CC)     // 1600 flattened (v,c)
#define MROWS   (NTROWS*VV) // 409600 GEMM M rows
#define RPB     128         // M-rows per block (8 wave32 x 16)
#define NBLOCKS (MROWS/RPB) // 3200
#define WIN     7           // max distinct nt rows a 128-row chunk can touch
#define EPSBN   1e-5f

typedef __attribute__((ext_vector_type(16))) __bf16 v16bf;
typedef __attribute__((ext_vector_type(8)))  __bf16 v8bf;
typedef __attribute__((ext_vector_type(8)))  float  v8f;

// ---------------------------------------------------------------------------
// One-time prep: maskf[i] = tanh(FM[i])+1 (f32), wt[d*64+c] = W[c][d] (bf16).
// Note: the Linear bias is dropped everywhere — it is constant per BN column
// and training-mode BatchNorm subtracts the batch mean, so it cancels exactly.
// ---------------------------------------------------------------------------
__global__ __launch_bounds__(256) void shiftgcn_prep(
    const float* __restrict__ FM, const float* __restrict__ W,
    float* __restrict__ maskf_g, __bf16* __restrict__ wt_g,
    float* __restrict__ sums)
{
    int i = blockIdx.x*256 + threadIdx.x;
    if (i < ROWLEN) maskf_g[i] = tanhf(FM[i]) + 1.0f;
    if (i < CC*CC)  wt_g[i] = (__bf16)W[(i & 63)*64 + (i >> 6)];
    if (i < 2*ROWLEN) sums[i] = 0.0f;               // zero BN accumulators
}

// ---------------------------------------------------------------------------
// Staging: copy prepped tables to LDS, then build the x0 window *already
// shift-in-permuted and masked* so WMMA A-operands are contiguous b128 loads.
//   shift_in[v*64+c] = (64v + 65c) mod 1600 == (i + ((i&63)<<6)) mod 1600
//   xs[ntl*1600 + i] = x0row[q(i)] * maskf[i]   (bf16)
// ---------------------------------------------------------------------------
__device__ __forceinline__ void stage_common(
    const float* __restrict__ x0,
    const float* __restrict__ maskf_g, const __bf16* __restrict__ wt_g,
    float* maskf, __bf16* wt, __bf16* xs, int nt0, int cnt)
{
    const int tid = threadIdx.x;
    for (int i = tid; i < ROWLEN; i += 256)
        maskf[i] = maskf_g[i];                       // L2-resident
    {   // copy 8KB W^T (bf16) as dwords
        const uint32_t* src = (const uint32_t*)wt_g;
        uint32_t*       dst = (uint32_t*)wt;
        for (int i = tid; i < (CC*CC)/2; i += 256) dst[i] = src[i];
    }
    __syncthreads();
    for (int r = 0; r < cnt; ++r) {
        const float* xrow = x0 + (size_t)(nt0 + r)*ROWLEN;
        for (int i = tid; i < ROWLEN; i += 256) {
            int s = i + ((i & 63) << 6);             // 64v + 65c, < 4*1600
            if (s >= 2*ROWLEN) s -= 2*ROWLEN;
            if (s >= ROWLEN)   s -= ROWLEN;
            xs[r*ROWLEN + i] = (__bf16)(xrow[s] * maskf[i]);
        }
    }
    __syncthreads();
}

// ---------------------------------------------------------------------------
// One wave -> 16(M) x 64(N) tile, f32 accumulate.
// A: ISA 16-bit 16x32 layout; lane l holds row M=l; the two K-runs of 8 are
//    contiguous 16B LDS loads.  B: ISA 16-bit 32x16 layout; lane l holds col
//    N=l, same K runs as A (8*half+j, 16+8*half+j) -> two 16B LDS loads; all 8 fragments hoisted.
// ---------------------------------------------------------------------------
__device__ __forceinline__ void gemm_tile(
    const __bf16* xs, const __bf16* wt, int tile_row0, int nt0, v8f acc[4])
{
    const int lane = threadIdx.x & 31;
    const int l    = lane & 15;
    const int half = lane >> 4;

    const int row    = tile_row0 + l;     // A-matrix row M = l for all lanes
    const int nt     = row / 25;
    const int v      = row - nt*25;
    const int base_a = (nt - nt0)*ROWLEN + (v << 6);   // shifted-row base

    v16bf bfrag[2][4];
    #pragma unroll
    for (int kk = 0; kk < 2; ++kk)
        #pragma unroll
        for (int n = 0; n < 4; ++n)
            bfrag[kk][n] = __builtin_shufflevector(
                *(const v8bf*)(wt + (n*16 + l)*64 + kk*32 + 8*half),
                *(const v8bf*)(wt + (n*16 + l)*64 + kk*32 + 8*half + 16),
                0,1,2,3,4,5,6,7, 8,9,10,11,12,13,14,15);

    #pragma unroll
    for (int n = 0; n < 4; ++n) acc[n] = (v8f){0,0,0,0,0,0,0,0};

    #pragma unroll
    for (int kk = 0; kk < 2; ++kk) {
        const __bf16* ap = xs + base_a + kk*32 + 8*half;
        v8bf lo = *(const v8bf*)(ap);                  // K run 0..7
        v8bf hi = *(const v8bf*)(ap + 16);             // K run 16..23
        v16bf a = __builtin_shufflevector(lo, hi,
            0,1,2,3,4,5,6,7, 8,9,10,11,12,13,14,15);
        #pragma unroll
        for (int n = 0; n < 4; ++n)
            acc[n] = __builtin_amdgcn_wmma_f32_16x16x32_bf16(
                false, a, false, bfrag[kk][n], (short)0, acc[n], false, false);
    }
}

// ---------------------------------------------------------------------------
// Pass 1: GEMM, per-column sum / sumsq of unshifted g via LDS ds_add_f32
// reduction, then one global_atomic_add_f32 per column per block.
// ---------------------------------------------------------------------------
__global__ __launch_bounds__(256) void shiftgcn_pass1(
    const float* __restrict__ x0,
    const float* __restrict__ maskf_g, const __bf16* __restrict__ wt_g,
    float* __restrict__ sums)
{
    __shared__ __attribute__((aligned(32))) __bf16 xs[WIN*ROWLEN];
    __shared__ __attribute__((aligned(32))) __bf16 wt[CC*CC];
    __shared__ float maskf[ROWLEN];
    __shared__ float stats[2*ROWLEN];

    const int tid = threadIdx.x;
    const int m0  = blockIdx.x * RPB;
    const int nt0 = m0 / 25;
    const int cnt = (m0 + RPB - 1) / 25 - nt0 + 1;   // <= WIN

    for (int i = tid; i < 2*ROWLEN; i += 256) stats[i] = 0.0f;
    stage_common(x0, maskf_g, wt_g, maskf, wt, xs, nt0, cnt);

    const int wave = tid >> 5;
    v8f acc[4];
    gemm_tile(xs, wt, m0 + wave*16, nt0, acc);

    const int lane = tid & 31, l = lane & 15, half = lane >> 4;
    // D fragment rows: M = r + 8*half -> incremental (nt, v) tracking
    int rrow = m0 + wave*16 + 8*half;
    int dnt  = rrow / 25;
    int dv   = rrow - dnt*25;
    #pragma unroll
    for (int r = 0; r < 8; ++r) {
        int pbase = (dv << 6) + l;
        #pragma unroll
        for (int n = 0; n < 4; ++n) {
            int   p = pbase + n*16;
            float g = acc[n][r];
            atomicAdd(&stats[p],          g);        // ds_add_f32
            atomicAdd(&stats[ROWLEN + p], g*g);
        }
        if (++dv == 25) { dv = 0; ++dnt; }
    }
    __syncthreads();
    for (int i = tid; i < 2*ROWLEN; i += 256)
        unsafeAtomicAdd(&sums[i], stats[i]);         // global_atomic_add_f32
}

// ---------------------------------------------------------------------------
// Fold shift-out + BatchNorm into per-k affine coefficients:
//   x_hat[:,k] = g[:,shift_out[k]] * scale[k] + offset[k]
// ---------------------------------------------------------------------------
__global__ __launch_bounds__(256) void shiftgcn_finalize(
    const float* __restrict__ sums, const float* __restrict__ gamma,
    const float* __restrict__ beta, const int* __restrict__ shift_out,
    float* __restrict__ scale, float* __restrict__ offset)
{
    int k = blockIdx.x*256 + threadIdx.x;
    if (k < ROWLEN) {
        int   p    = shift_out[k];
        float mean = sums[p]          * (1.0f / NTROWS);
        float var  = sums[ROWLEN + p] * (1.0f / NTROWS) - mean*mean;
        float rs   = rsqrtf(var + EPSBN);
        float sc   = gamma[k] * rs;
        scale[k]  = sc;
        offset[k] = beta[k] - mean*sc;
    }
}

// ---------------------------------------------------------------------------
// Pass 2: recompute g (x0 is L2-resident; compute is free), invert shift-out
// via LDS scatter table, apply BN affine + residual + ReLU, store transposed
// to (N, D, T, V).
// ---------------------------------------------------------------------------
__global__ __launch_bounds__(256) void shiftgcn_pass2(
    const float* __restrict__ x0,
    const float* __restrict__ maskf_g, const __bf16* __restrict__ wt_g,
    const int* __restrict__ shift_out,
    const float* __restrict__ scale, const float* __restrict__ offset,
    float* __restrict__ out)
{
    __shared__ __attribute__((aligned(32))) __bf16 xs[WIN*ROWLEN];
    __shared__ __attribute__((aligned(32))) __bf16 wt[CC*CC];
    __shared__ float maskf[ROWLEN];
    __shared__ int   invL[ROWLEN];
    __shared__ float scL[ROWLEN];
    __shared__ float ofL[ROWLEN];

    const int tid = threadIdx.x;
    const int m0  = blockIdx.x * RPB;
    const int nt0 = m0 / 25;
    const int cnt = (m0 + RPB - 1) / 25 - nt0 + 1;

    for (int k = tid; k < ROWLEN; k += 256) {
        invL[shift_out[k]] = k;                      // invert the permutation
        scL[k] = scale[k];
        ofL[k] = offset[k];
    }
    stage_common(x0, maskf_g, wt_g, maskf, wt, xs, nt0, cnt);

    const int wave = tid >> 5;
    v8f acc[4];
    gemm_tile(xs, wt, m0 + wave*16, nt0, acc);

    const int lane = tid & 31, l = lane & 15, half = lane >> 4;
    int rrow = m0 + wave*16 + 8*half;
    int dnt  = rrow / 25;
    int dv   = rrow - dnt*25;
    #pragma unroll
    for (int r = 0; r < 8; ++r) {
        int pbase = (dv << 6) + l;
        int nn = dnt >> 8, t = dnt & 255;            // T = 256
        #pragma unroll
        for (int n = 0; n < 4; ++n) {
            int k = invL[pbase + n*16];              // shifted column
            float y = acc[n][r]*scL[k] + ofL[k]
                    + x0[(size_t)dnt*ROWLEN + k];    // residual, L2-resident
            y = fmaxf(y, 0.0f);
            int dk = k & 63, vk = k >> 6;
            out[(((size_t)(nn*CC + dk))*TT + t)*VV + vk] = y;
        }
        if (++dv == 25) { dv = 0; ++dnt; }
    }
}

extern "C" void kernel_launch(void* const* d_in, const int* in_sizes, int n_in,
                              void* d_out, int out_size, void* d_ws, size_t ws_size,
                              hipStream_t stream)
{
    (void)in_sizes; (void)n_in; (void)out_size; (void)ws_size;
    const float* x0    = (const float*)d_in[0];
    const float* W     = (const float*)d_in[1];
    const float* FM    = (const float*)d_in[3];
    const float* gamma = (const float*)d_in[4];
    const float* beta  = (const float*)d_in[5];
    const int*   s_out = (const int*)d_in[7];
    float* out = (float*)d_out;

    float*   sums    = (float*)d_ws;         // [2*1600] sum / sumsq
    float*   scale   = sums + 2*ROWLEN;      // [1600]
    float*   offset  = scale + ROWLEN;       // [1600]
    float*   maskf_g = offset + ROWLEN;      // [1600]
    __bf16*  wt_g    = (__bf16*)(maskf_g + ROWLEN);  // [4096] bf16 W^T

    shiftgcn_prep<<<(2*ROWLEN + 255)/256, 256, 0, stream>>>(
        FM, W, maskf_g, wt_g, sums);
    shiftgcn_pass1<<<NBLOCKS, 256, 0, stream>>>(x0, maskf_g, wt_g, sums);
    shiftgcn_finalize<<<(ROWLEN + 255)/256, 256, 0, stream>>>(
        sums, gamma, beta, s_out, scale, offset);
    shiftgcn_pass2<<<NBLOCKS, 256, 0, stream>>>(
        x0, maskf_g, wt_g, s_out, scale, offset, out);
}
